// MambaMock_56324201120365
// MI455X (gfx1250) — matrix operand order repaired and checked by reading, not compiled, not run on hardware
//
#include <hip/hip_runtime.h>
#include <hip/hip_bf16.h>

typedef __attribute__((ext_vector_type(16))) __bf16 v16bf;
typedef __attribute__((ext_vector_type(8)))  __bf16 v8bf;
typedef __attribute__((ext_vector_type(8)))  float  v8f;

#define DMODEL 512
#define BATCH  8
#define SEQ    4096
#define NG     1536          // 3*D
#define LN_EPS 1e-5f

#define GWG   16             // recurrence workgroups (split of hidden dim)
#define DG    32             // hidden channels per workgroup
#define NLOC  96             // 3*DG rows of W_hh owned per workgroup

#define MSUB  8              // M-subtiles per wave in projection GEMM

// ---------------------------------------------------------------------------
// fp32 -> bf16 pre-conversion (each thread packs 8 elements).
// ---------------------------------------------------------------------------
__global__ __launch_bounds__(256) void cvt_bf16_kernel(
    const float* __restrict__ src, __bf16* __restrict__ dst, int n8)
{
  const int i = blockIdx.x * blockDim.x + threadIdx.x;
  if (i >= n8) return;
  const size_t base = (size_t)i * 8;
  const float4 a = *(const float4*)(src + base);
  const float4 b = *(const float4*)(src + base + 4);
  v8bf o;
  o[0] = (__bf16)a.x; o[1] = (__bf16)a.y; o[2] = (__bf16)a.z; o[3] = (__bf16)a.w;
  o[4] = (__bf16)b.x; o[5] = (__bf16)b.y; o[6] = (__bf16)b.z; o[7] = (__bf16)b.w;
  *(v8bf*)(dst + base) = o;
}

// ---------------------------------------------------------------------------
// Kernel A: x_proj[m, :] = x[m, :] @ W_ih^T + b_ih   (M=32768, K=512, N=1536)
// bf16 inputs; one wave computes a 128x16 output strip (8 M-subtiles reuse
// one B fragment per K-chunk).
// ---------------------------------------------------------------------------
__global__ __launch_bounds__(256) void gru_xproj_kernel(
    const __bf16* __restrict__ xb, const __bf16* __restrict__ wb,
    const float* __restrict__ b_ih, float* __restrict__ xproj)
{
  const int lane  = threadIdx.x & 31;
  const int wave  = threadIdx.x >> 5;
  const int gw    = blockIdx.x * 8 + wave;   // 24576 waves total
  const int mg    = gw / 96;                 // 256 groups of 128 rows
  const int nt    = gw % 96;                 // 96 column tiles
  const int tile_m = mg * (16 * MSUB);
  const int tile_n = nt * 16;

  const int mrow  = lane & 15;               // row within 16-tile
  const int khalf = lane >> 4;               // A: K {0..15} vs {16..31}
  const int ncol  = tile_n + (lane & 15);    // B/C column

  v8f acc[MSUB] = {};

  for (int kc = 0; kc < DMODEL; kc += 32) {
    // B tile: B[k][n] = W_ih[n][k]; 16 consecutive bf16 at fixed n (32B aligned)
    const v16bf bt = *(const v16bf*)(wb + (size_t)ncol * DMODEL + kc + khalf * 16);

#pragma unroll
    for (int t = 0; t < MSUB; t++) {
      const __bf16* ar = xb + (size_t)(tile_m + t * 16 + mrow) * DMODEL + kc + khalf * 16;
      const v8bf a0 = *(const v8bf*)(ar);        // K +0..7   (16B aligned)
      const v8bf a1 = *(const v8bf*)(ar + 8);    // K +8..15
      const v16bf at = __builtin_shufflevector(
          a0, a1, 0, 1, 2, 3, 4, 5, 6, 7, 8, 9, 10, 11, 12, 13, 14, 15);
      acc[t] = __builtin_amdgcn_wmma_f32_16x16x32_bf16(
          false, at, false, bt, (short)0, acc[t], false, false);
    }
  }

  const float bias = b_ih[ncol];
#pragma unroll
  for (int t = 0; t < MSUB; t++) {
    const int rbase = tile_m + t * 16 + khalf * 8;   // C: vgpr i -> row i + 8*(lane>=16)
#pragma unroll
    for (int i = 0; i < 8; i++)
      xproj[(size_t)(rbase + i) * NG + ncol] = acc[t][i] + bias;
  }
}

// ---------------------------------------------------------------------------
// Init: zero h ping-pong buffers and grid-sync words (ws is poisoned 0xAA).
// ---------------------------------------------------------------------------
__global__ void gru_init_kernel(float* hbuf, unsigned* sync)
{
  const int tid = blockIdx.x * blockDim.x + threadIdx.x;
  for (int i = tid; i < 2 * BATCH * DMODEL; i += blockDim.x * gridDim.x) hbuf[i] = 0.0f;
  if (tid < 4) sync[tid] = 0u;
}

// ---------------------------------------------------------------------------
// Kernel B: persistent GRU recurrence. 16 workgroups, each owns 32 hidden
// channels (96 rows of W_hh, bf16-resident in LDS). Per step:
//   hp = h @ W_slice^T via v_wmma_f32_16x16x32_bf16 (M padded 8->16, vector
//   ds_load_b128 fragment loads), fp32 gates, atomic grid barrier.
// ---------------------------------------------------------------------------
__global__ __launch_bounds__(256) void gru_recur_kernel(
    const float* __restrict__ xproj, const float* __restrict__ W_hh,
    const float* __restrict__ b_hh, float* __restrict__ hbuf,
    float* __restrict__ hseq, unsigned* __restrict__ sync_cnt,
    unsigned* __restrict__ sync_gen)
{
  __shared__ __bf16 w_lds[NLOC * DMODEL];        // 96 KB
  __shared__ __bf16 h_lds[16 * DMODEL];          // 16 KB (rows 8..15 = zero pad)
  __shared__ float  hp_lds[NLOC * BATCH];        // 3 KB
  __shared__ float  b_lds[NLOC];

  const int g    = blockIdx.x;                   // 0..15: hidden-dim slice
  const int tid  = threadIdx.x;
  const int lane = tid & 31;
  const int wave = tid >> 5;

  // One-time: W_hh slice -> LDS bf16, b_hh slice, zero pad rows of h_lds.
  for (int idx = tid; idx < NLOC * DMODEL; idx += 256) {
    const int nloc = idx >> 9;                   // / 512
    const int k    = idx & (DMODEL - 1);
    const int grow = (nloc >> 5) * DMODEL + g * DG + (nloc & 31);
    w_lds[idx] = (__bf16)W_hh[(size_t)grow * DMODEL + k];
  }
  for (int idx = tid; idx < NLOC; idx += 256)
    b_lds[idx] = b_hh[(idx >> 5) * DMODEL + g * DG + (idx & 31)];
  for (int idx = tid; idx < 8 * DMODEL; idx += 256)
    h_lds[8 * DMODEL + idx] = (__bf16)0.0f;
  __syncthreads();

  const int mrow  = lane & 15;
  const int khalf = lane >> 4;

  for (int l = 0; l < SEQ; ++l) {
    const float* hin  = hbuf + (l & 1) * (BATCH * DMODEL);
    float*       hout = hbuf + ((l + 1) & 1) * (BATCH * DMODEL);

    // phase 1: broadcast h into LDS as bf16
    for (int idx = tid; idx < BATCH * DMODEL; idx += 256)
      h_lds[idx] = (__bf16)hin[idx];
    __syncthreads();

    // phase 2: hp tile per wave (waves 0..5 cover 96 local columns)
    if (wave < 6) {
      const int tile_n = wave * 16;
      v8f acc = {};
      for (int kc = 0; kc < DMODEL; kc += 32) {
        const v16bf bt = *(const v16bf*)(
            w_lds + (size_t)(tile_n + (lane & 15)) * DMODEL + kc + khalf * 16);
        const __bf16* ar = h_lds + (size_t)mrow * DMODEL + kc + khalf * 16;
        const v8bf a0 = *(const v8bf*)(ar);
        const v8bf a1 = *(const v8bf*)(ar + 8);
        const v16bf at = __builtin_shufflevector(
            a0, a1, 0, 1, 2, 3, 4, 5, 6, 7, 8, 9, 10, 11, 12, 13, 14, 15);
        acc = __builtin_amdgcn_wmma_f32_16x16x32_bf16(
            false, at, false, bt, (short)0, acc, false, false);
      }
      // valid batch rows m=0..7 live in lanes 0..15, vgprs 0..7
      if (khalf == 0) {
        const int nloc = tile_n + (lane & 15);
#pragma unroll
        for (int i = 0; i < 8; i++) hp_lds[nloc * BATCH + i] = acc[i];
      }
    }
    __syncthreads();

    // phase 3: gates + h_new (256 threads = 8 batch x 32 channels)
    {
      const int m    = tid >> 5;
      const int dloc = tid & 31;
      const int dglb = g * DG + dloc;
      const size_t xrow = (size_t)(m * SEQ + l) * NG;
      const float xr = xproj[xrow + dglb];
      const float xz = xproj[xrow + DMODEL + dglb];
      const float xn = xproj[xrow + 2 * DMODEL + dglb];
      const float hr = hp_lds[(0 * DG + dloc) * BATCH + m] + b_lds[0 * DG + dloc];
      const float hz = hp_lds[(1 * DG + dloc) * BATCH + m] + b_lds[1 * DG + dloc];
      const float hn = hp_lds[(2 * DG + dloc) * BATCH + m] + b_lds[2 * DG + dloc];
      const float hp = hin[m * DMODEL + dglb];
      const float r  = 1.0f / (1.0f + __expf(-(xr + hr)));
      const float z  = 1.0f / (1.0f + __expf(-(xz + hz)));
      const float n  = tanhf(xn + r * hn);
      const float hnew = (1.0f - z) * n + z * hp;
      hout[m * DMODEL + dglb] = hnew;
      hseq[(size_t)(m * SEQ + l) * DMODEL + dglb] = hnew;
    }

    // grid barrier: all 16 WGs finish step l before anyone starts l+1
    __threadfence();
    __syncthreads();
    if (tid == 0) {
      const unsigned gen =
          __hip_atomic_load(sync_gen, __ATOMIC_ACQUIRE, __HIP_MEMORY_SCOPE_AGENT);
      const unsigned prev =
          __hip_atomic_fetch_add(sync_cnt, 1u, __ATOMIC_ACQ_REL, __HIP_MEMORY_SCOPE_AGENT);
      if (prev == GWG - 1) {
        __hip_atomic_store(sync_cnt, 0u, __ATOMIC_RELAXED, __HIP_MEMORY_SCOPE_AGENT);
        __hip_atomic_fetch_add(sync_gen, 1u, __ATOMIC_RELEASE, __HIP_MEMORY_SCOPE_AGENT);
      } else {
        while (__hip_atomic_load(sync_gen, __ATOMIC_ACQUIRE, __HIP_MEMORY_SCOPE_AGENT) == gen)
          __builtin_amdgcn_s_sleep(1);
      }
    }
    __syncthreads();
  }
}

// ---------------------------------------------------------------------------
// Kernel C: LayerNorm over D=512 per row; one wave per row, shfl reductions.
// ---------------------------------------------------------------------------
__global__ __launch_bounds__(256) void gru_ln_kernel(
    const float* __restrict__ hseq, const float* __restrict__ gamma,
    const float* __restrict__ beta, float* __restrict__ out)
{
  const int lane = threadIdx.x & 31;
  const int wave = threadIdx.x >> 5;
  const size_t row = (size_t)blockIdx.x * 8 + wave;   // 32768 rows
  const float* rp = hseq + row * DMODEL;

  float v[16];
  float s = 0.0f, s2 = 0.0f;
#pragma unroll
  for (int j = 0; j < 16; j++) {
    v[j] = rp[j * 32 + lane];
    s  += v[j];
    s2 += v[j] * v[j];
  }
#pragma unroll
  for (int off = 16; off > 0; off >>= 1) {
    s  += __shfl_xor(s,  off, 32);
    s2 += __shfl_xor(s2, off, 32);
  }
  const float mean = s * (1.0f / DMODEL);
  const float var  = s2 * (1.0f / DMODEL) - mean * mean;
  const float inv  = rsqrtf(var + LN_EPS);

  float* op = out + row * DMODEL;
#pragma unroll
  for (int j = 0; j < 16; j++) {
    const int d = j * 32 + lane;
    op[d] = (v[j] - mean) * inv * gamma[d] + beta[d];
  }
}

// ---------------------------------------------------------------------------
extern "C" void kernel_launch(void* const* d_in, const int* in_sizes, int n_in,
                              void* d_out, int out_size, void* d_ws, size_t ws_size,
                              hipStream_t stream)
{
  const float* x     = (const float*)d_in[0];
  const float* W_ih  = (const float*)d_in[1];
  const float* W_hh  = (const float*)d_in[2];
  const float* b_ih  = (const float*)d_in[3];
  const float* b_hh  = (const float*)d_in[4];
  const float* gamma = (const float*)d_in[5];
  const float* beta  = (const float*)d_in[6];
  float* out = (float*)d_out;

  char* ws = (char*)d_ws;
  size_t off = 0;
  float* xproj = (float*)(ws + off);  off += (size_t)32768 * 1536 * 4;   // 192 MB
  float* hseq  = (float*)(ws + off);  off += (size_t)32768 * 512 * 4;    //  64 MB
  __bf16* xb   = (__bf16*)(ws + off); off += (size_t)32768 * 512 * 2;    //  32 MB
  __bf16* wb   = (__bf16*)(ws + off); off += (size_t)1536 * 512 * 2;     // 1.5 MB
  float* hbuf  = (float*)(ws + off);  off += (size_t)2 * BATCH * DMODEL * 4;
  unsigned* sync = (unsigned*)(ws + off);

  const int nx8 = (32768 * 512) / 8;   // 2097152
  const int nw8 = (1536 * 512) / 8;    // 98304

  gru_init_kernel<<<1, 256, 0, stream>>>(hbuf, sync);
  cvt_bf16_kernel<<<(nx8 + 255) / 256, 256, 0, stream>>>(x, xb, nx8);
  cvt_bf16_kernel<<<(nw8 + 255) / 256, 256, 0, stream>>>(W_ih, wb, nw8);
  gru_xproj_kernel<<<24576 / 8, 256, 0, stream>>>(xb, wb, b_ih, xproj);
  gru_recur_kernel<<<GWG, 256, 0, stream>>>(xproj, W_hh, b_hh, hbuf, hseq,
                                            sync, sync + 1);
  gru_ln_kernel<<<4096, 256, 0, stream>>>(hseq, gamma, beta, out);
}
